// SpikingLMUFFTCell_79216376807577
// MI455X (gfx1250) — matrix operand order repaired and checked by reading, not compiled, not run on hardware
//
#include <hip/hip_runtime.h>
#include <hip/hip_bf16.h>

typedef __attribute__((ext_vector_type(16))) _Float16 v16h;
typedef __attribute__((ext_vector_type(8)))  _Float16 v8h;
typedef __attribute__((ext_vector_type(8)))  float    v8f;

namespace {
constexpr int kB = 64;        // batch
constexpr int kV = 1024;      // sequence length (time)
constexpr int kC = 384;       // input channels
constexpr int kM = 256;       // LMU memory size
constexpr int kH = 768;       // hidden units
constexpr int kF = kM + kC;   // 640, concat feature dim
constexpr float kEps = 1e-5f;
}

// WMMA A-operand K index for 16-bit A 16x32 (ISA 7.12.2):
// lanes 0-15: e=0..7 -> K=e, e=8..15 -> K=e+8 ; hi lanes add 8.
__device__ __forceinline__ int kmapA(int e, int hi) { return e + (hi << 3) + (e & 8); }
// Contiguous K index this code uses for both A and B fragments: lanes 0-15 K=e, lanes 16-31 K=e+16.
__device__ __forceinline__ int kmapB(int e, int hi) { return e + (hi << 4); }

__device__ __forceinline__ v8f wmma16x16x32(const v16h& a, const v16h& b, const v8f& c) {
  return __builtin_amdgcn_wmma_f32_16x16x32_f16(false, a, false, b, (short)0, c, false, false);
}

// Async 16B memory -> LDS copy (per lane), tracked by ASYNCcnt.
__device__ __forceinline__ void async_ld_b128(unsigned lds_off, unsigned long long gaddr) {
  asm volatile("global_load_async_to_lds_b128 %0, %1, off" :: "v"(lds_off), "v"(gaddr) : "memory");
}
__device__ __forceinline__ void async_wait0() {
  asm volatile("s_wait_asynccnt 0" ::: "memory");
}
// Low 32 bits of a generic pointer to __shared__ = LDS byte offset (SHARED_BASE aperture in high bits).
__device__ __forceinline__ unsigned lds_off32(const void* p) {
  return (unsigned)(size_t)p;
}

// ---------------- conversion / packing kernels ----------------

__global__ void cvt_f32_to_f16(const float* __restrict__ src, _Float16* __restrict__ dst, int n) {
  for (int i = blockIdx.x * blockDim.x + threadIdx.x; i < n; i += gridDim.x * blockDim.x)
    dst[i] = (_Float16)src[i];
}

// H [kM][kV] f32 -> Hp packed into WMMA B-fragment order:
// tile (dt: K=32 chunk, qt: N=16 chunk) -> per lane, 16 contiguous halves:
// Hp[(((dt*16)+qt)*32 + lane)*16 + e] = H[q0+lane&15][d0 + e + 16*(lane>>4)]
__global__ void pack_Hp(const float* __restrict__ H, _Float16* __restrict__ Hp) {
  int i = blockIdx.x * blockDim.x + threadIdx.x;
  if (i >= kV * kM) return;
  int e    = i & 15;
  int lane = (i >> 4) & 31;
  int qt   = (i >> 9) & 15;
  int dt   = i >> 13;                       // 0..31
  int d = (dt << 5) + e + ((lane >> 4) << 4);
  int q = (qt << 4) + (lane & 15);
  Hp[i] = (_Float16)H[q * kV + d];
}

// W_h [kH][kF] f32 -> Wp packed into WMMA B-fragment order (ft: K=32 chunk of 640, ot: N=16 chunk of 768).
__global__ void pack_Wp(const float* __restrict__ W, _Float16* __restrict__ Wp) {
  int i = blockIdx.x * blockDim.x + threadIdx.x;
  if (i >= kF * kH) return;
  int e    = i & 15;
  int lane = (i >> 4) & 31;
  int tile = i >> 9;                        // ft*48 + ot
  int ot = tile % 48, ft = tile / 48;
  int f = (ft << 5) + e + ((lane >> 4) << 4);
  int o = (ot << 4) + (lane & 15);
  Wp[i] = (_Float16)W[o * kF + f];
}

// ---------------- u path: GEMV + stats + LIF scan ----------------

__global__ __launch_bounds__(256) void upre_kernel(const float* __restrict__ x,
    const float* __restrict__ Wuw, const float* __restrict__ Wub,
    float* __restrict__ u_pre) {
  __shared__ float sw[kC];
  for (int i = threadIdx.x; i < kC; i += blockDim.x) sw[i] = Wuw[i];
  __syncthreads();
  int i = blockIdx.x * blockDim.x + threadIdx.x;   // (b*kV + t)
  const float* xr = x + (size_t)i * kC;
  float s = Wub[0];
  for (int c = 0; c < kC; ++c) s += xr[c] * sw[c];
  u_pre[i] = s;
}

// One block per channel c: reduce over B*V, write mean and inv_std.
// Layout: pre[((b*C + c)*kV) + t]
__global__ __launch_bounds__(256) void stats_kernel(const float* __restrict__ pre,
    float* __restrict__ stats, int C) {
  int c = blockIdx.x;
  __shared__ float s1[256], s2[256];
  float a = 0.f, b2 = 0.f;
  const int N = kB * kV;
  for (int i = threadIdx.x; i < N; i += 256) {
    int bb = i >> 10, t = i & (kV - 1);
    float v = pre[((size_t)bb * C + c) * kV + t];
    a += v; b2 += v * v;
  }
  s1[threadIdx.x] = a; s2[threadIdx.x] = b2;
  __syncthreads();
  for (int w = 128; w > 0; w >>= 1) {
    if (threadIdx.x < w) { s1[threadIdx.x] += s1[threadIdx.x + w]; s2[threadIdx.x] += s2[threadIdx.x + w]; }
    __syncthreads();
  }
  if (threadIdx.x == 0) {
    float mu  = s1[0] / (float)N;
    float var = s2[0] / (float)N - mu * mu;   // biased var, matches reference
    stats[2 * c] = mu;
    stats[2 * c + 1] = rsqrtf(var + kEps);
  }
}

// LIF: v = (v + x)/2 ; s = (v >= 1) ; v *= (1-s). Spikes written as f16 {0,1}.
// us row = [1024 zeros | 1024 spikes] -> causal Toeplitz reads need no branch.
__global__ void lif_u_kernel(const float* __restrict__ u_pre, const float* __restrict__ stats,
    const float* __restrict__ g, const float* __restrict__ be, _Float16* __restrict__ us) {
  int b = threadIdx.x;
  if (b >= kB) return;
  _Float16* row = us + (size_t)b * 2 * kV;
  for (int t = 0; t < kV; ++t) row[t] = (_Float16)0.f;
  float mu = stats[0], is = stats[1], ga = g[0], bb = be[0];
  float v = 0.f;
  for (int t = 0; t < kV; ++t) {
    float xn = ga * (u_pre[b * kV + t] - mu) * is + bb;
    v = 0.5f * (v + xn);
    float sp = (v >= 1.0f) ? 1.0f : 0.0f;
    row[kV + t] = (_Float16)sp;
    v = (sp != 0.0f) ? 0.0f : v;
  }
}

// ---------------- causal conv as implicit-Toeplitz WMMA GEMM ----------------
// m_pre[b,q,t] = sum_d H[q,d] * u[b,t-d]   (K loop limited by causality)
// Block: one (b, t-tile); 4 waves x 4 q-tiles = 16 q-tiles (all of kM=256).
__global__ __launch_bounds__(128) void conv_wmma_kernel(const _Float16* __restrict__ us,
    const _Float16* __restrict__ Hp, float* __restrict__ m_pre) {
  __shared__ __align__(16) _Float16 sU[2 * kV];
  int b  = blockIdx.x >> 6;
  int t0 = (blockIdx.x & 63) << 4;
  {
    // async-stage the zero-padded spike row (4 KB) into LDS: 128 lanes x 16B x 2 passes
    unsigned lbase = lds_off32(sU);
    unsigned long long gbase = (unsigned long long)(size_t)(us + (size_t)b * 2 * kV);
    #pragma unroll
    for (int p = 0; p < 2; ++p) {
      unsigned off = threadIdx.x * 16u + p * 2048u;
      async_ld_b128(lbase + off, gbase + off);
    }
    async_wait0();
  }
  __syncthreads();

  int lane = threadIdx.x & 31;
  int wave = threadIdx.x >> 5;
  int nrow = lane & 15;
  int hi   = lane >> 4;

  const v16h* Bp = (const v16h*)Hp;

  v8f acc[4];
  #pragma unroll
  for (int j = 0; j < 4; ++j)
    #pragma unroll
    for (int r = 0; r < 8; ++r) acc[j][r] = 0.0f;

  for (int d0 = 0; d0 < t0 + 16; d0 += 32) {
    int dt = d0 >> 5;
    v16h a;
    #pragma unroll
    for (int e = 0; e < 16; ++e)
      a[e] = sU[kV + t0 + nrow - (d0 + kmapB(e, hi))];   // zero-pad handles d > t
    if (d0 + 32 < t0 + 16)
      __builtin_prefetch((const void*)&Bp[(((dt + 1) << 4) + (wave << 2)) * 32 + lane], 0, 0);
    #pragma unroll
    for (int j = 0; j < 4; ++j) {
      int qt = (wave << 2) + j;
      v16h bm = Bp[(size_t)(((dt << 4) + qt) * 32 + lane)];  // one 32B fragment load
      acc[j] = wmma16x16x32(a, bm, acc[j]);
    }
  }
  #pragma unroll
  for (int j = 0; j < 4; ++j) {
    int q = (((wave << 2) + j) << 4) + nrow;
    float* out = m_pre + ((size_t)b * kM + q) * kV + t0 + (hi << 3);
    #pragma unroll
    for (int r = 0; r < 8; ++r) out[r] = acc[j][r];
  }
}

// LIF over t per (b,q); writes spikes transposed to [b][t][q] for the next GEMM's A tiles.
__global__ __launch_bounds__(256) void lif_m_kernel(const float* __restrict__ m_pre,
    const float* __restrict__ stats, const float* __restrict__ g, const float* __restrict__ be,
    _Float16* __restrict__ msp) {
  int b = blockIdx.x, q = threadIdx.x;
  float mu = stats[2 * q], is = stats[2 * q + 1], ga = g[q], bb = be[q];
  const float* src = m_pre + ((size_t)b * kM + q) * kV;
  float v = 0.f;
  for (int t = 0; t < kV; ++t) {
    float xn = ga * (src[t] - mu) * is + bb;
    v = 0.5f * (v + xn);
    float sp = (v >= 1.0f) ? 1.0f : 0.0f;
    msp[((size_t)b * kV + t) * kM + q] = (_Float16)sp;
    v = (sp != 0.0f) ? 0.0f : v;
  }
}

// ---------------- hidden GEMM: h_pre[b,o,t] = concat(msp,x)[b,t,:] . W_h[o,:] ----------------
// Block: one (b, t-tile). Async-stage 16x640 f16 activations in LDS, reuse over 48 o-tiles.
__global__ __launch_bounds__(128) void gemm_h_wmma_kernel(const _Float16* __restrict__ msp,
    const _Float16* __restrict__ xh, const _Float16* __restrict__ Wp,
    const float* __restrict__ Whb, float* __restrict__ h_pre) {
  constexpr int LD = 656;                       // padded row stride (halves)
  __shared__ __align__(16) _Float16 sA[16 * LD];
  int b  = blockIdx.x >> 6;
  int t0 = (blockIdx.x & 63) << 4;
  {
    unsigned lbase = lds_off32(sA);
    for (int i = threadIdx.x; i < 16 * 80; i += blockDim.x) {  // 80x16B per row (32 msp + 48 x)
      int r = i / 80, j = i % 80;
      size_t row = (size_t)b * kV + t0 + r;
      const _Float16* src = (j < 32) ? (msp + row * kM + j * 8)
                                     : (xh  + row * kC + (j - 32) * 8);
      async_ld_b128(lbase + (unsigned)(r * LD + j * 8) * 2u,
                    (unsigned long long)(size_t)src);
    }
    async_wait0();
  }
  __syncthreads();

  int lane = threadIdx.x & 31;
  int wave = threadIdx.x >> 5;
  int nrow = lane & 15;
  int hi   = lane >> 4;

  const v16h* Wv = (const v16h*)Wp;

  for (int jt = wave; jt < kH / 16; jt += 4) {
    int o0 = jt << 4;
    v8f acc;
    #pragma unroll
    for (int r = 0; r < 8; ++r) acc[r] = 0.0f;
    for (int f0 = 0; f0 < kF; f0 += 32) {
      int ft = f0 >> 5;
      // A fragment: two aligned 16B runs from LDS + register shuffle
      const v8h* pa = (const v8h*)(sA + nrow * LD + f0 + (hi << 4));
      v8h c0 = pa[0];
      v8h c1 = pa[1];
      v16h a = __builtin_shufflevector(c0, c1, 0, 1, 2, 3, 4, 5, 6, 7,
                                               8, 9, 10, 11, 12, 13, 14, 15);
      if (f0 + 32 < kF)
        __builtin_prefetch((const void*)&Wv[((ft + 1) * 48 + jt) * 32 + lane], 0, 0);
      v16h bm = Wv[(size_t)((ft * 48 + jt) * 32 + lane)];   // one 32B fragment load
      acc = wmma16x16x32(a, bm, acc);
    }
    int o = o0 + nrow;
    float bias = Whb[o];
    float* out = h_pre + ((size_t)b * kH + o) * kV + t0 + (hi << 3);
    #pragma unroll
    for (int r = 0; r < 8; ++r) out[r] = acc[r] + bias;
  }
}

// LIF over t per (b,o); writes h [B,V,HID] and h_n [B,HID] (f32 outputs).
__global__ __launch_bounds__(256) void lif_h_kernel(const float* __restrict__ h_pre,
    const float* __restrict__ stats, const float* __restrict__ g, const float* __restrict__ be,
    float* __restrict__ h_out, float* __restrict__ hn_out) {
  int gid = blockIdx.x * blockDim.x + threadIdx.x;
  int b = gid / kH, o = gid % kH;
  if (b >= kB) return;
  float mu = stats[2 * o], is = stats[2 * o + 1], ga = g[o], bb = be[o];
  const float* src = h_pre + ((size_t)b * kH + o) * kV;
  float v = 0.f, sp = 0.f;
  for (int t = 0; t < kV; ++t) {
    float xn = ga * (src[t] - mu) * is + bb;
    v = 0.5f * (v + xn);
    sp = (v >= 1.0f) ? 1.0f : 0.0f;
    h_out[((size_t)b * kV + t) * kH + o] = sp;
    v = (sp != 0.0f) ? 0.0f : v;
  }
  hn_out[(size_t)b * kH + o] = sp;
}

// ---------------- launcher ----------------

extern "C" void kernel_launch(void* const* d_in, const int* in_sizes, int n_in,
                              void* d_out, int out_size, void* d_ws, size_t ws_size,
                              hipStream_t stream) {
  (void)in_sizes; (void)n_in; (void)out_size; (void)ws_size;
  const float* x    = (const float*)d_in[0];
  const float* Wuw  = (const float*)d_in[1];
  const float* Wub  = (const float*)d_in[2];
  const float* Whw  = (const float*)d_in[3];
  const float* Whb  = (const float*)d_in[4];
  const float* bnug = (const float*)d_in[5];
  const float* bnub = (const float*)d_in[6];
  const float* bnmg = (const float*)d_in[7];
  const float* bnmb = (const float*)d_in[8];
  const float* bnhg = (const float*)d_in[9];
  const float* bnhb = (const float*)d_in[10];
  const float* H    = (const float*)d_in[11];

  char* ws = (char*)d_ws;
  size_t off = 0;
  auto take = [&](size_t bytes) { char* p = ws + off; off += (bytes + 255) & ~(size_t)255; return p; };
  _Float16* xh    = (_Float16*)take((size_t)kB * kV * kC * 2);   // x in f16
  _Float16* Hp    = (_Float16*)take((size_t)kV * kM * 2);        // H packed to B-fragment order
  _Float16* Wp    = (_Float16*)take((size_t)kF * kH * 2);        // W_h packed to B-fragment order
  float*    u_pre = (float*)   take((size_t)kB * kV * 4);
  _Float16* us    = (_Float16*)take((size_t)kB * 2 * kV * 2);    // zero-padded spikes
  float*    m_pre = (float*)   take((size_t)kB * kM * kV * 4);
  _Float16* msp   = (_Float16*)take((size_t)kB * kV * kM * 2);
  float*    h_pre = (float*)   take((size_t)kB * kH * kV * 4);
  float*    st_u  = (float*)   take(2 * 4);
  float*    st_m  = (float*)   take((size_t)2 * kM * 4);
  float*    st_h  = (float*)   take((size_t)2 * kH * 4);

  float* h_out  = (float*)d_out;
  float* hn_out = h_out + (size_t)kB * kV * kH;

  cvt_f32_to_f16<<<4096, 256, 0, stream>>>(x, xh, kB * kV * kC);
  pack_Hp<<<(kV * kM + 255) / 256, 256, 0, stream>>>(H, Hp);
  pack_Wp<<<(kF * kH + 255) / 256, 256, 0, stream>>>(Whw, Wp);

  upre_kernel<<<kB * kV / 256, 256, 0, stream>>>(x, Wuw, Wub, u_pre);
  stats_kernel<<<1, 256, 0, stream>>>(u_pre, st_u, 1);
  lif_u_kernel<<<1, 64, 0, stream>>>(u_pre, st_u, bnug, bnub, us);

  conv_wmma_kernel<<<kB * (kV / 16), 128, 0, stream>>>(us, Hp, m_pre);
  stats_kernel<<<kM, 256, 0, stream>>>(m_pre, st_m, kM);
  lif_m_kernel<<<kB, kM, 0, stream>>>(m_pre, st_m, bnmg, bnmb, msp);

  gemm_h_wmma_kernel<<<kB * (kV / 16), 128, 0, stream>>>(msp, xh, Wp, Whb, h_pre);
  stats_kernel<<<kH, 256, 0, stream>>>(h_pre, st_h, kH);
  lif_h_kernel<<<kB * kH / 256, 256, 0, stream>>>(h_pre, st_h, bnhg, bnhb, h_out, hn_out);
}
